// GNNL1V2ImpBS_9251359556303
// MI455X (gfx1250) — matrix operand order repaired and checked by reading, not compiled, not run on hardware
//
#include <hip/hip_runtime.h>
#include <hip/hip_bf16.h>

typedef __attribute__((ext_vector_type(16))) __bf16 v16bf;
typedef __attribute__((ext_vector_type(8)))  __bf16 v8bf;
typedef __attribute__((ext_vector_type(4)))  __bf16 v4bf;
typedef __attribute__((ext_vector_type(8)))  float  v8f;
typedef __attribute__((ext_vector_type(4)))  float  v4f;
typedef __attribute__((ext_vector_type(4)))  int    v4i;

// ---------------------------------------------------------------------------
// CDNA5 async global->LDS copy (ASYNCcnt path), with safe fallback.
// Probe result: builtin expects (v4i AS1*, v4i AS3*, imm offset, imm cpol).
// ---------------------------------------------------------------------------
#if defined(__has_builtin)
#  if __has_builtin(__builtin_amdgcn_global_load_async_to_lds_b128) && \
      __has_builtin(__builtin_amdgcn_s_wait_asynccnt)
#    define HAVE_ASYNC_LDS 1
#  endif
#endif
#ifndef HAVE_ASYNC_LDS
#  define HAVE_ASYNC_LDS 0
#endif

__device__ __forceinline__ void copy16_g2l(const __bf16* g, __bf16* l) {
#if HAVE_ASYNC_LDS
    __builtin_amdgcn_global_load_async_to_lds_b128(
        (__attribute__((address_space(1))) v4i*)g,
        (__attribute__((address_space(3))) v4i*)l, 0, 0);
#else
    *(v8bf*)l = *(const v8bf*)g;
#endif
}

template<int N>
__device__ __forceinline__ void wait_async() {
#if HAVE_ASYNC_LDS
    __builtin_amdgcn_s_wait_asynccnt(N);
#endif
}

// ---------------------------------------------------------------------------
// Tiled bf16 WMMA GEMM: C(MxN,f32) = A(MxK,bf16) * Bt(NxK,bf16)^T [+ bias]
// Block tile 128x64, BK=32 (one v_wmma_f32_16x16x32_bf16 K-step), 8 waves,
// wave (mw,nw) owns 32x32 (2x2 WMMA frags). Double-buffered LDS staging via
// async global->LDS b128 copies (3 per wave per tile).
// Requires M%128==0, N%64==0, K%32==0, lda/ldbt%8==0 (all true here).
// ---------------------------------------------------------------------------
template<bool HAS_BIAS>
__global__ __launch_bounds__(256)
void gemm_bf16_wmma(const __bf16* __restrict__ A, const __bf16* __restrict__ Bt,
                    float* __restrict__ C, const float* __restrict__ bias,
                    int M, int N, int K, int lda, int ldbt, int ldc)
{
    constexpr int BM = 128, BN = 64, BK = 32, LS = 40;   // LS: LDS row stride (bf16)
    __shared__ __bf16 As[2][BM * LS];
    __shared__ __bf16 Bs[2][BN * LS];

    const int tid  = threadIdx.x;
    const int wave = tid >> 5, lane = tid & 31;
    const int hi   = lane >> 4, lo  = lane & 15;
    const int mw   = wave & 3,  nw  = wave >> 2;          // 4 x 2 wave grid
    const int tileM = blockIdx.x * BM, tileN = blockIdx.y * BN;

    // A tile: 128 rows x 32 bf16 = 512 16B-chunks -> 2 per thread.
    const int ar0 = tid >> 2,        akc = (tid & 3) * 8;   // chunk 0: rows 0..63
    const int ar1 = (tid + 256) >> 2;                        // chunk 1: rows 64..127
    // B tile: 64 rows x 32 bf16 = 256 16B-chunks -> 1 per thread.
    const int bn  = tid >> 2,        bkc = (tid & 3) * 8;

    auto stage = [&](int buf, int k0) {
        copy16_g2l(A  + (size_t)(tileM + ar0) * lda  + k0 + akc, &As[buf][ar0 * LS + akc]);
        copy16_g2l(A  + (size_t)(tileM + ar1) * lda  + k0 + akc, &As[buf][ar1 * LS + akc]);
        copy16_g2l(Bt + (size_t)(tileN + bn)  * ldbt + k0 + bkc, &Bs[buf][bn  * LS + bkc]);
    };

    v8f acc[2][2] = {};
    const int nk = K / BK;

    stage(0, 0);
    for (int kt = 0; kt < nk; ++kt) {
        const int cur = kt & 1;
        if (kt + 1 < nk) { stage(cur ^ 1, (kt + 1) * BK); wait_async<3>(); }
        else             { wait_async<0>(); }
        __syncthreads();

        // A 16x32 frag: lane(hi,lo): row=lo, elem j -> K = 16*(j>>3)+8*hi+(j&7)
        v16bf af[2];
        #pragma unroll
        for (int mi = 0; mi < 2; ++mi) {
            int row = mw * 32 + mi * 16 + lo;
            v8bf x0 = *(const v8bf*)(&As[cur][row * LS + 8 * hi]);
            v8bf x1 = *(const v8bf*)(&As[cur][row * LS + 16 + 8 * hi]);
            #pragma unroll
            for (int j = 0; j < 8; ++j) { af[mi][j] = x0[j]; af[mi][j + 8] = x1[j]; }
        }
        // B 32x16 frag: lane(hi,lo): col=lo, elem j -> K = 16*(j>>3)+8*hi+(j&7)
        v16bf bfrag[2];
        #pragma unroll
        for (int ni = 0; ni < 2; ++ni) {
            int col = nw * 32 + ni * 16 + lo;
            v8bf x0 = *(const v8bf*)(&Bs[cur][col * LS + 8 * hi]);
            v8bf x1 = *(const v8bf*)(&Bs[cur][col * LS + 16 + 8 * hi]);
            #pragma unroll
            for (int j = 0; j < 8; ++j) { bfrag[ni][j] = x0[j]; bfrag[ni][j + 8] = x1[j]; }
        }

        #pragma unroll
        for (int mi = 0; mi < 2; ++mi)
            #pragma unroll
            for (int ni = 0; ni < 2; ++ni)
                acc[mi][ni] = __builtin_amdgcn_wmma_f32_16x16x32_bf16(
                    false, af[mi], false, bfrag[ni], (short)0, acc[mi][ni], false, false);

        __syncthreads();
    }

    // Epilogue. C/D layout: VGPR r -> row = r + 8*hi, col = lo.
    #pragma unroll
    for (int ni = 0; ni < 2; ++ni) {
        const int col = tileN + nw * 32 + ni * 16 + lo;
        float bv = 0.0f;
        if constexpr (HAS_BIAS) bv = bias[col];
        #pragma unroll
        for (int mi = 0; mi < 2; ++mi) {
            #pragma unroll
            for (int r = 0; r < 8; ++r) {
                const int row = tileM + mw * 32 + mi * 16 + r + 8 * hi;
                if constexpr (HAS_BIAS) C[(size_t)row * ldc + col] = acc[mi][ni][r] + bv;
                else                    C[(size_t)row * ldc + col] = acc[mi][ni][r];
            }
        }
    }
}

// ---------------------------------------------------------------------------
// GAT attention over N=9 nodes, D=256. One block per (b, head); thread t owns
// feature d=t. f1/f2 via wave32 shfl_xor + LDS reduce; leaky-relu + adj mask +
// softmax + optional importance; out = att @ Wh; ELU; bf16 store.
// ---------------------------------------------------------------------------
template<bool USE_IMP>
__global__ __launch_bounds__(256)
void gat_attn(const float* __restrict__ Wh, int whStride,
              const float* __restrict__ a,          // 512 floats per head
              const float* __restrict__ adj,        // 9x9
              const float* __restrict__ imp,        // (B,9,9) or nullptr
              __bf16* __restrict__ out, int outStride)
{
    const int b = blockIdx.x, h = blockIdx.y;
    const int t = threadIdx.x;
    const int wave = t >> 5, lane = t & 31;

    const float* whb = Wh + (size_t)b * 9 * whStride + (size_t)h * 256;
    const float* av  = a + (size_t)h * 512;
    const float a1 = av[t], a2 = av[256 + t];

    float wh[9];
    #pragma unroll
    for (int n = 0; n < 9; ++n) wh[n] = whb[(size_t)n * whStride + t];

    float p[18];
    #pragma unroll
    for (int n = 0; n < 9; ++n) { p[n] = wh[n] * a1; p[9 + n] = wh[n] * a2; }

    #pragma unroll
    for (int off = 16; off > 0; off >>= 1)
        #pragma unroll
        for (int k = 0; k < 18; ++k)
            p[k] += __shfl_xor(p[k], off, 32);

    __shared__ float red[8][18];
    __shared__ float f[18];
    __shared__ float adjs[81];
    __shared__ float imps[81];

    if (lane == 0)
        #pragma unroll
        for (int k = 0; k < 18; ++k) red[wave][k] = p[k];
    if (t < 81) {
        adjs[t] = adj[t];
        if (USE_IMP) imps[t] = imp[(size_t)b * 81 + t];
    }
    __syncthreads();
    if (t < 18) {
        float s = 0.0f;
        #pragma unroll
        for (int w = 0; w < 8; ++w) s += red[w][t];
        f[t] = s;
    }
    __syncthreads();

    __bf16* ob = out + (size_t)b * 9 * outStride + (size_t)h * 256;
    #pragma unroll
    for (int i = 0; i < 9; ++i) {
        float e[9];
        float m = -3.0e38f;
        #pragma unroll
        for (int j = 0; j < 9; ++j) {
            float v = f[i] + f[9 + j];
            v = (v > 0.0f) ? v : 0.2f * v;              // leaky_relu(0.2)
            if (!(adjs[i * 9 + j] > 0.0f)) v = -9.0e15f;
            e[j] = v;
            m = fmaxf(m, v);
        }
        float s = 0.0f;
        #pragma unroll
        for (int j = 0; j < 9; ++j) { e[j] = __expf(e[j] - m); s += e[j]; }
        const float inv = 1.0f / s;
        float acc = 0.0f;
        #pragma unroll
        for (int j = 0; j < 9; ++j) {
            float aij = e[j] * inv;
            if (USE_IMP) aij *= imps[i * 9 + j];
            acc += aij * wh[j];
        }
        acc = (acc > 0.0f) ? acc : (__expf(acc) - 1.0f); // ELU
        ob[(size_t)i * outStride + t] = (__bf16)acc;
    }
}

// ---------------------------------------------------------------------------
// f32 -> bf16 flat convert (vectorized, 4 elems/thread).
// ---------------------------------------------------------------------------
__global__ __launch_bounds__(256)
void cvt_f32_bf16(const float* __restrict__ s, __bf16* __restrict__ d, long n4)
{
    long i = (long)blockIdx.x * 256 + threadIdx.x;
    if (i < n4) {
        v4f v = *(const v4f*)(s + i * 4);
        v4bf b;
        #pragma unroll
        for (int j = 0; j < 4; ++j) b[j] = (__bf16)v[j];
        *(v4bf*)(d + i * 4) = b;
    }
}

// ---------------------------------------------------------------------------
// Transpose + convert: src f32 (K x N) -> dst bf16 (N x K). K,N % 32 == 0.
// ---------------------------------------------------------------------------
__global__ __launch_bounds__(256)
void transpose_cvt(const float* __restrict__ src, __bf16* __restrict__ dst,
                   int K, int N)
{
    __shared__ float tile[32][33];
    const int kb = blockIdx.x * 32, nb = blockIdx.y * 32;
    const int tx = threadIdx.x & 31, ty = threadIdx.x >> 5;   // ty 0..7
    #pragma unroll
    for (int i = ty; i < 32; i += 8)
        tile[i][tx] = src[(size_t)(kb + i) * N + nb + tx];
    __syncthreads();
    #pragma unroll
    for (int i = ty; i < 32; i += 8)
        dst[(size_t)(nb + i) * K + kb + tx] = (__bf16)tile[tx][i];
}

// ---------------------------------------------------------------------------
// Launcher
// ---------------------------------------------------------------------------
extern "C" void kernel_launch(void* const* d_in, const int* in_sizes, int n_in,
                              void* d_out, int out_size, void* d_ws, size_t ws_size,
                              hipStream_t stream) {
    const float* x      = (const float*)d_in[0];
    const float* adj    = (const float*)d_in[1];
    const float* imp    = (const float*)d_in[2];
    const float* Wheads = (const float*)d_in[3];
    const float* aheads = (const float*)d_in[4];
    const float* Wout   = (const float*)d_in[5];
    const float* aout   = (const float*)d_in[6];
    const float* linw   = (const float*)d_in[7];
    const float* linb   = (const float*)d_in[8];
    float* out = (float*)d_out;

    const int B = 8192, H = 4;
    const int BR = B * 9;                                    // 73728 rows

    // Workspace layout (bytes):
    char* p = (char*)d_ws;
    __bf16* x_bf  = (__bf16*)p;  p += (size_t)BR * 1024 * 2;   // 151 MB  (reused: Wh2 f32)
    float*  Wh    = (float*)p;   p += (size_t)BR * 1024 * 4;   // 302 MB  (reused: h2_bf)
    __bf16* h1_bf = (__bf16*)p;  p += (size_t)BR * 1024 * 2;   // 151 MB
    __bf16* WhT   = (__bf16*)p;  p += (size_t)H * 256 * 1024 * 2;   // W_heads^T per head
    __bf16* WoT   = (__bf16*)p;  p += (size_t)256 * 1024 * 2;       // W_out^T
    __bf16* lw_bf = (__bf16*)p;                                      // lin_w (already NxK)
    float*  Wh2   = (float*)x_bf;                               // (73728,256) f32
    __bf16* h2_bf = (__bf16*)Wh;                                // (8192,2304) bf16

    dim3 blk(256);

    // --- precision/layout prep (bf16 everywhere the WMMAs eat) ---
    cvt_f32_bf16<<<dim3((BR * 1024 / 4 + 255) / 256), blk, 0, stream>>>(
        x, x_bf, (long)BR * 1024 / 4);
    for (int h = 0; h < H; ++h)
        transpose_cvt<<<dim3(1024 / 32, 256 / 32), blk, 0, stream>>>(
            Wheads + (size_t)h * 1024 * 256, WhT + (size_t)h * 256 * 1024, 1024, 256);
    transpose_cvt<<<dim3(1024 / 32, 256 / 32), blk, 0, stream>>>(Wout, WoT, 1024, 256);
    cvt_f32_bf16<<<dim3((256 * 2304 / 4 + 255) / 256), blk, 0, stream>>>(
        linw, lw_bf, (long)256 * 2304 / 4);

    // GEMM1: per-head x @ W_heads[h] -> Wh columns [h*256, (h+1)*256)
    for (int h = 0; h < H; ++h)
        gemm_bf16_wmma<false><<<dim3(BR / 128, 256 / 64), blk, 0, stream>>>(
            x_bf, WhT + (size_t)h * 256 * 1024, Wh + (size_t)h * 256, nullptr,
            BR, 256, 1024, 1024, 1024, 1024);

    // Stage 2: multi-head attention + importance + ELU -> h1_bf (73728,1024)
    gat_attn<true><<<dim3(B, H), blk, 0, stream>>>(Wh, 1024, aheads, adj, imp, h1_bf, 1024);

    // GEMM2: h1 @ W_out -> Wh2 (73728,256)
    gemm_bf16_wmma<false><<<dim3(BR / 128, 256 / 64), blk, 0, stream>>>(
        h1_bf, WoT, Wh2, nullptr, BR, 256, 1024, 1024, 1024, 256);

    // Stage 4: single-head attention (no importance) + ELU -> h2_bf (8192,2304)
    gat_attn<false><<<dim3(B, 1), blk, 0, stream>>>(Wh2, 256, aout, adj, nullptr, h2_bf, 256);

    // GEMM3: h2(8192,2304) @ lin_w.T + lin_b -> out (8192,256)
    gemm_bf16_wmma<true><<<dim3(B / 128, 256 / 64), blk, 0, stream>>>(
        h2_bf, lw_bf, out, linb, B, 256, 2304, 2304, 2304, 256);
}
